// ExplicitInterClassGraphLoss_55319178772552
// MI455X (gfx1250) — compile-verified, matrix operand order checked by reading, not run on hardware
//
#include <hip/hip_runtime.h>
#include <math.h>

typedef __attribute__((ext_vector_type(16))) _Float16 v16h;
typedef __attribute__((ext_vector_type(8)))  _Float16 v8h;
typedef __attribute__((ext_vector_type(4)))  _Float16 v4h;
typedef __attribute__((ext_vector_type(8)))  float    v8f;
typedef __attribute__((ext_vector_type(4)))  float    v4f;

#define NROWS 8192
#define KDIM  512
#define NBIN  40
// pi/2 / (NBIN-1)
#define ZSTEP 0.04027682889217683f

#define MACRO 256                 // macro-tile edge (rows and cols)
#define GRID1D (NROWS / MACRO)    // 32
#define NBLOCKS ((GRID1D * (GRID1D + 1)) / 2)   // 528

// ---------------------------------------------------------------------------
// Kernel 1: row-normalize fp32 features, emit f16 matrix into workspace.
// One block (128 threads) per row; each thread handles 4 contiguous floats.
// ---------------------------------------------------------------------------
__global__ __launch_bounds__(128) void normalize_f16_kernel(
    const float* __restrict__ feat, _Float16* __restrict__ fn)
{
    __shared__ float red[128];
    const int row = blockIdx.x;
    const int tid = threadIdx.x;

    const float* rp = feat + (size_t)row * KDIM + tid * 4;
    v4f v = *(const v4f*)rp;
    red[tid] = v.x * v.x + v.y * v.y + v.z * v.z + v.w * v.w;
    __syncthreads();
#pragma unroll
    for (int off = 64; off > 0; off >>= 1) {
        if (tid < off) red[tid] += red[tid + off];
        __syncthreads();
    }
    const float inv = 1.0f / fmaxf(sqrtf(red[0]), 1e-8f);

    v4h o;
    o.x = (_Float16)(v.x * inv);
    o.y = (_Float16)(v.y * inv);
    o.z = (_Float16)(v.z * inv);
    o.w = (_Float16)(v.w * inv);
    *(v4h*)(fn + (size_t)row * KDIM + tid * 4) = o;
}

// ---------------------------------------------------------------------------
// Kernel 2: fused Gram-matrix + loss over one 256x256 macro-tile.
// Upper-triangular macro-tile enumeration; off-diagonal blocks weighted 2x.
// 512 threads = 16 waves arranged 4x4; each wave computes a 64x64 region
// = 4x4 tiles of 16x16 (16 accumulators), K-stepped by 32 with
// v_wmma_f32_16x16x32_f16. 16 b128 loads feed 16 WMMAs per K-step.
// ---------------------------------------------------------------------------
__global__ __launch_bounds__(512) void gram_loss_kernel(
    const _Float16* __restrict__ fn, const int* __restrict__ label,
    float* __restrict__ partial)
{
    __shared__ float sLab[512];   // [0..255] row labels, [256..511] col labels (as float)
    __shared__ float sRed[512];

    // Decode linear block id -> (by, bx) with bx >= by over GRID1D x GRID1D.
    int rem = blockIdx.x;
    int by = 0;
    int span = GRID1D;
    while (rem >= span) { rem -= span; ++by; --span; }
    const int bx = by + rem;

    const int rowBase = by * MACRO;
    const int colBase = bx * MACRO;

    const int tid = threadIdx.x;
    sLab[tid] = (float)((tid < 256) ? label[rowBase + tid]
                                    : label[colBase + (tid - 256)]);
    __syncthreads();

    const int lane = tid & 31;
    const int w    = tid >> 5;        // 0..15
    const int wRow = w >> 2;          // 0..3 : 64-row slab
    const int wCol = w & 3;           // 0..3 : 64-col slab

    const int lr = lane & 15;         // row/col within 16-tile
    const int hi = lane >> 4;         // K-half selector (A/B frag layout)

    const int aRow0 = rowBase + wRow * 64;
    const int bCol0 = colBase + wCol * 64;

    v8f acc[16] = {};

    for (int kk = 0; kk < KDIM; kk += 32) {
        const int kch = kk + (hi << 3);   // this lane's first K chunk
        v16h a[4], b[4];
#pragma unroll
        for (int i = 0; i < 4; ++i) {
            const _Float16* p = fn + (size_t)(aRow0 + i * 16 + lr) * KDIM + kch;
            v8h lo = *(const v8h*)(p);
            v8h hv = *(const v8h*)(p + 16);
            a[i] = __builtin_shufflevector(lo, hv, 0,1,2,3,4,5,6,7,8,9,10,11,12,13,14,15);
        }
#pragma unroll
        for (int j = 0; j < 4; ++j) {
            const _Float16* p = fn + (size_t)(bCol0 + j * 16 + lr) * KDIM + kch;
            v8h lo = *(const v8h*)(p);
            v8h hv = *(const v8h*)(p + 16);
            b[j] = __builtin_shufflevector(lo, hv, 0,1,2,3,4,5,6,7,8,9,10,11,12,13,14,15);
        }
#pragma unroll
        for (int i = 0; i < 4; ++i)
#pragma unroll
            for (int j = 0; j < 4; ++j)
                acc[i * 4 + j] = __builtin_amdgcn_wmma_f32_16x16x32_f16(
                    false, a[i], false, b[j], (short)0, acc[i * 4 + j],
                    false, false);
    }

    // Epilogue: D layout — VGPR r holds (M = r + 8*hi, N = lane&15).
    float lsum = 0.0f;
#pragma unroll
    for (int i = 0; i < 4; ++i) {
        const int mBase = wRow * 64 + i * 16 + (hi << 3);
#pragma unroll
        for (int j = 0; j < 4; ++j) {
            const int nIdx = wCol * 64 + j * 16 + lr;
            const float lj = sLab[256 + nIdx];
            v8f c = acc[i * 4 + j];
#pragma unroll
            for (int r = 0; r < 8; ++r) {
                const float li = sLab[mBase + r];
                float lbl = __cosf(fabsf(li - lj) * ZSTEP);
                lbl = fminf(fmaxf(lbl, 0.0f), 1.0f);
                const float d = c[r] - lbl;
                lsum += d * d;
            }
        }
    }
    if (bx != by) lsum *= 2.0f;   // account for mirrored block

    sRed[tid] = lsum;
    __syncthreads();
#pragma unroll
    for (int off = 256; off > 0; off >>= 1) {
        if (tid < off) sRed[tid] += sRed[tid + off];
        __syncthreads();
    }
    if (tid == 0) partial[blockIdx.x] = sRed[0];
}

// ---------------------------------------------------------------------------
// Kernel 3: deterministic final reduction -> mean.
// ---------------------------------------------------------------------------
__global__ __launch_bounds__(256) void final_reduce_kernel(
    const float* __restrict__ partial, int n, float* __restrict__ out)
{
    __shared__ float sRed[256];
    float s = 0.0f;
    for (int i = threadIdx.x; i < n; i += 256) s += partial[i];
    sRed[threadIdx.x] = s;
    __syncthreads();
#pragma unroll
    for (int off = 128; off > 0; off >>= 1) {
        if (threadIdx.x < off) sRed[threadIdx.x] += sRed[threadIdx.x + off];
        __syncthreads();
    }
    if (threadIdx.x == 0)
        out[0] = sRed[0] * (1.0f / ((float)NROWS * (float)NROWS));
}

// ---------------------------------------------------------------------------
extern "C" void kernel_launch(void* const* d_in, const int* in_sizes, int n_in,
                              void* d_out, int out_size, void* d_ws, size_t ws_size,
                              hipStream_t stream)
{
    (void)in_sizes; (void)n_in; (void)out_size; (void)ws_size;

    const int*   label = (const int*)d_in[0];   // batch_label (8192,) int32
    const float* feat  = (const float*)d_in[1]; // batch_feat (8192,512) fp32

    _Float16* fn      = (_Float16*)d_ws;                              // 8 MB
    float*    partial = (float*)((char*)d_ws + (size_t)NROWS * KDIM * sizeof(_Float16));

    normalize_f16_kernel<<<NROWS, 128, 0, stream>>>(feat, fn);

    gram_loss_kernel<<<NBLOCKS, 512, 0, stream>>>(fn, label, partial);

    final_reduce_kernel<<<1, 256, 0, stream>>>(partial, NBLOCKS, (float*)d_out);
}
